// MambaSequenceClassifier_1743756722396
// MI455X (gfx1250) — matrix operand order repaired and checked by reading, not compiled, not run on hardware
//
#include <hip/hip_runtime.h>

// ---------------------------------------------------------------------------
// Mamba sequence classifier for MI455X (gfx1250), wave32 + WMMA f16->f32.
// GEMMs run on v_wmma_f32_16x16x32_f16 (fp16 in, fp32 accumulate), 64x64
// output tile per wave (16 v8f accumulators) for 1:1 b128-load:wmma ratio.
// Selective scan keeps per-channel state in fp32 VGPRs (8192 parallel chans).
// ---------------------------------------------------------------------------

#define B_    8
#define L_    2048
#define DM    512          // d_model
#define NL    4            // layers
#define DS    16           // d_state
#define DC    4            // d_conv
#define DI    1024         // d_inner
#define DTR   32           // dt_rank
#define NC    64           // classes
#define MTOT  (B_ * L_)    // 16384 rows

typedef __attribute__((ext_vector_type(16))) _Float16 v16h;
typedef __attribute__((ext_vector_type(8)))  _Float16 v8h;
typedef __attribute__((ext_vector_type(8)))  float    v8f;

// ---------------------------------------------------------------------------
// Helpers
// ---------------------------------------------------------------------------
__device__ __forceinline__ float sigmoidf_(float x) { return 1.f / (1.f + __expf(-x)); }

__device__ __forceinline__ v8f wmma32(v16h a, v16h b, v8f c) {
  // D = A(16x32) * B(32x16) + C, fp32 accumulate (probe-verified builtin)
  return __builtin_amdgcn_wmma_f32_16x16x32_f16(
      /*neg_a=*/false, a, /*neg_b=*/false, b,
      /*c_mod=*/(short)0, c, /*reuse_a=*/false, /*reuse_b=*/false);
}

// A fragment (16x32, M x K), row-major source, ld = K.
// lanes0-15 hold M=0..15 with K 0..15 contiguous; lanes16-31 same M with
// K 16..31 -> two 16-byte loads per lane.
__device__ __forceinline__ v16h frag_a(const _Float16* __restrict__ A, int ld,
                                       int lane, int k) {
  int row = lane & 15;
  int kb  = k + ((lane >> 4) << 4);
  const _Float16* q = A + (size_t)row * ld + kb;
  v8h lo = *(const v8h*)(q);
  v8h hi = *(const v8h*)(q + 8);
  v16h r;
#pragma unroll
  for (int i = 0; i < 8; ++i) { r[i] = lo[i]; r[i + 8] = hi[i]; }
  return r;
}

// B fragment (32x16, K x N) sourced from W[N,K] row-major (B[k][n] = W[n][k]).
// Contiguous k order (same as frag_a): lanes0-15 = col n with K 0..15 contiguous, lanes16-31 K 16..31:
// one contiguous 32-byte load per lane.
__device__ __forceinline__ v16h frag_b(const _Float16* __restrict__ W, int ld,
                                       int lane, int k) {
  int col = lane & 15;
  int kb  = k + ((lane >> 4) << 4);
  return *(const v16h*)(W + (size_t)col * ld + kb);
}

// ---------------------------------------------------------------------------
// fp32 -> fp16 conversion (weights)
// ---------------------------------------------------------------------------
__global__ void k_cvt_f16(const float* __restrict__ src,
                          _Float16* __restrict__ dst, int n) {
  int i = blockIdx.x * 256 + threadIdx.x;
  if (i < n) dst[i] = (_Float16)src[i];
}

// ---------------------------------------------------------------------------
// Embedding gather
// ---------------------------------------------------------------------------
__global__ void k_embed(const int* __restrict__ x, const float* __restrict__ emb,
                        float* __restrict__ h) {
  int i = blockIdx.x * 256 + threadIdx.x;
  if (i >= MTOT * DM) return;
  int row = i >> 9;          // / DM
  int d   = i & (DM - 1);
  h[i] = emb[(size_t)x[row] * DM + d];
}

// ---------------------------------------------------------------------------
// LayerNorm of fp32 residual -> fp16 activations (one block per row)
// ---------------------------------------------------------------------------
__global__ __launch_bounds__(256)
void k_ln_f16(const float* __restrict__ h, const float* __restrict__ w,
              const float* __restrict__ bias, _Float16* __restrict__ out) {
  __shared__ float ws0[8], ws1[8], mrs[2];
  int row = blockIdx.x;
  const float* hp = h + (size_t)row * DM;
  float s = 0.f, s2 = 0.f;
  for (int i = threadIdx.x; i < DM; i += 256) {
    float v = hp[i]; s += v; s2 += v * v;
  }
  for (int o = 16; o > 0; o >>= 1) {
    s  += __shfl_down(s, o);
    s2 += __shfl_down(s2, o);
  }
  int lane = threadIdx.x & 31, wid = threadIdx.x >> 5;
  if (lane == 0) { ws0[wid] = s; ws1[wid] = s2; }
  __syncthreads();
  if (threadIdx.x == 0) {
    float a = 0.f, b2 = 0.f;
#pragma unroll
    for (int i = 0; i < 8; ++i) { a += ws0[i]; b2 += ws1[i]; }
    float mu  = a * (1.f / DM);
    float var = b2 * (1.f / DM) - mu * mu;
    mrs[0] = mu; mrs[1] = rsqrtf(var + 1e-5f);
  }
  __syncthreads();
  float mu = mrs[0], rs = mrs[1];
  for (int i = threadIdx.x; i < DM; i += 256)
    out[(size_t)row * DM + i] = (_Float16)((hp[i] - mu) * rs * w[i] + bias[i]);
}

// ---------------------------------------------------------------------------
// Final LN statistics (mu, rstd per row)
// ---------------------------------------------------------------------------
__global__ __launch_bounds__(256)
void k_lnstats(const float* __restrict__ h, float* __restrict__ stats) {
  __shared__ float ws0[8], ws1[8];
  int row = blockIdx.x;
  const float* hp = h + (size_t)row * DM;
  float s = 0.f, s2 = 0.f;
  for (int i = threadIdx.x; i < DM; i += 256) {
    float v = hp[i]; s += v; s2 += v * v;
  }
  for (int o = 16; o > 0; o >>= 1) {
    s  += __shfl_down(s, o);
    s2 += __shfl_down(s2, o);
  }
  int lane = threadIdx.x & 31, wid = threadIdx.x >> 5;
  if (lane == 0) { ws0[wid] = s; ws1[wid] = s2; }
  __syncthreads();
  if (threadIdx.x == 0) {
    float a = 0.f, b2 = 0.f;
#pragma unroll
    for (int i = 0; i < 8; ++i) { a += ws0[i]; b2 += ws1[i]; }
    float mu  = a * (1.f / DM);
    float var = b2 * (1.f / DM) - mu * mu;
    stats[row * 2 + 0] = mu;
    stats[row * 2 + 1] = rsqrtf(var + 1e-5f);
  }
}

// ---------------------------------------------------------------------------
// WMMA GEMM: C[M,N] (op) A[M,K] * W[N,K]^T.  64x64 tile per wave,
// 8 waves per block; M % 64 == 0, N % 64 == 0, K % 32 == 0.
// mode: 0 = f32 store, 1 = f32 +=, 2 = f16 store, 3 = softplus(acc+bias) f32
// ---------------------------------------------------------------------------
__global__ __launch_bounds__(256)
void k_gemm_wmma(const _Float16* __restrict__ A, const _Float16* __restrict__ W,
                 void* __restrict__ C, const float* __restrict__ bias,
                 int M, int N, int K, int mode) {
  int wave = (blockIdx.x << 3) + (threadIdx.x >> 5);
  int lane = threadIdx.x & 31;
  int tn   = N >> 6;                 // 64-col tiles
  int m0   = (wave / tn) << 6;       // 64 rows per wave tile
  int n0   = (wave % tn) << 6;

  const _Float16* At[4];
  const _Float16* Wt[4];
#pragma unroll
  for (int i = 0; i < 4; ++i) {
    At[i] = A + (size_t)(m0 + 16 * i) * K;
    Wt[i] = W + (size_t)(n0 + 16 * i) * K;
  }

  v8f acc[4][4];
#pragma unroll
  for (int i = 0; i < 4; ++i)
#pragma unroll
    for (int j = 0; j < 4; ++j) acc[i][j] = (v8f){};

  for (int k = 0; k < K; k += 32) {
    if (k + 128 < K) {  // WGP-scope global_prefetch_b8 (locality 3 -> near$)
      __builtin_prefetch(At[0] + (size_t)(lane & 15) * K + k + 128, 0, 3);
      __builtin_prefetch(Wt[0] + (size_t)(lane & 15) * K + k + 128, 0, 3);
    }
    v16h a[4], b[4];
#pragma unroll
    for (int i = 0; i < 4; ++i) a[i] = frag_a(At[i], K, lane, k);
#pragma unroll
    for (int j = 0; j < 4; ++j) b[j] = frag_b(Wt[j], K, lane, k);
#pragma unroll
    for (int i = 0; i < 4; ++i)
#pragma unroll
      for (int j = 0; j < 4; ++j)
        acc[i][j] = wmma32(a[i], b[j], acc[i][j]);
  }

  // Epilogue. C/D layout: vgpr i -> row (lane<16 ? i : 8+i), col = lane&15.
  float*    Cf = (float*)C;
  _Float16* Ch = (_Float16*)C;
  int rbase = (lane >> 4) << 3;
  int col   = lane & 15;
#pragma unroll
  for (int mi = 0; mi < 4; ++mi) {
#pragma unroll
    for (int j = 0; j < 4; ++j) {
      int gr = m0 + mi * 16 + rbase;
      int gc = n0 + j * 16 + col;
#pragma unroll
      for (int i = 0; i < 8; ++i) {
        size_t off = (size_t)(gr + i) * N + gc;
        float v = acc[mi][j][i];
        if (mode == 0)      Cf[off] = v;
        else if (mode == 1) Cf[off] += v;
        else if (mode == 2) Ch[off] = (_Float16)v;
        else {
          float t = v + bias[gc];                         // softplus(t)
          Cf[off] = (t > 20.f) ? t : __logf(1.f + __expf(t));
        }
      }
    }
  }
}

// ---------------------------------------------------------------------------
// Causal depthwise conv (width 4) + SiLU over xs half of xz (f16 in/out)
// ---------------------------------------------------------------------------
__global__ void k_conv_silu(const _Float16* __restrict__ xz,
                            const float* __restrict__ cw,
                            const float* __restrict__ cb,
                            _Float16* __restrict__ out) {
  int i = blockIdx.x * 256 + threadIdx.x;
  if (i >= MTOT * DI) return;
  int d   = i & (DI - 1);
  int row = i >> 10;             // b*L + l
  int l   = row & (L_ - 1);
  float acc = cb[d];
#pragma unroll
  for (int j = 0; j < DC; ++j) {
    int ls = l - (DC - 1) + j;
    if (ls >= 0)
      acc += (float)xz[(size_t)(row - (DC - 1) + j) * (2 * DI) + d] * cw[d * DC + j];
  }
  out[i] = (_Float16)(acc * sigmoidf_(acc));
}

// ---------------------------------------------------------------------------
// Slice dt columns [0,32) of x_dbl (stride 64) into a dense f16 matrix
// ---------------------------------------------------------------------------
__global__ void k_slice_dt(const float* __restrict__ xdbl,
                           _Float16* __restrict__ dt16) {
  int i = blockIdx.x * 256 + threadIdx.x;
  if (i >= MTOT * DTR) return;
  int r = i & (DTR - 1), row = i >> 5;
  dt16[i] = (_Float16)xdbl[(size_t)row * 64 + r];
}

// ---------------------------------------------------------------------------
// Selective scan: one thread per (b, d) channel, fp32 state in registers.
// Fuses +u*D and *silu(z) gating; writes f16 for the out_proj WMMA GEMM.
// ---------------------------------------------------------------------------
__global__ __launch_bounds__(256)
void k_scan(const _Float16* __restrict__ u16, const float* __restrict__ delta,
            const float* __restrict__ xdbl, const _Float16* __restrict__ xz,
            const float* __restrict__ Alog, const float* __restrict__ Dv,
            _Float16* __restrict__ y16) {
  int t = blockIdx.x * 256 + threadIdx.x;   // [0, B_*DI)
  int b = t >> 10;
  int d = t & (DI - 1);
  float a[DS], st[DS];
#pragma unroll
  for (int n = 0; n < DS; ++n) {
    a[n]  = -__expf(Alog[d * DS + n]);
    st[n] = 0.f;
  }
  float Dd = Dv[d];
  size_t rb = (size_t)b * L_;
  for (int l = 0; l < L_; ++l) {
    size_t row = rb + l;
    float u  = (float)u16[row * DI + d];
    float dt = delta[row * DI + d];
    const float* xb = xdbl + row * 64;      // [dt(32) | B(16) | C(16)]
    float dtu = dt * u;
    float y = 0.f;
#pragma unroll
    for (int n = 0; n < DS; ++n) {
      float dA = __expf(dt * a[n]);
      st[n] = fmaf(dA, st[n], dtu * xb[32 + n]);
      y     = fmaf(st[n], xb[48 + n], y);
    }
    float z  = (float)xz[row * (2 * DI) + DI + d];
    float yo = (y + u * Dd) * (z * sigmoidf_(z));
    y16[row * DI + d] = (_Float16)yo;
  }
}

// ---------------------------------------------------------------------------
// Mean-pool of final LayerNorm: pooled[b,d] = w[d]*mean_l((h-mu)*rstd) + b[d]
// ---------------------------------------------------------------------------
__global__ void k_pool(const float* __restrict__ h, const float* __restrict__ stats,
                       const float* __restrict__ w, const float* __restrict__ bias,
                       float* __restrict__ pooled) {
  int t = blockIdx.x * 256 + threadIdx.x;   // [0, B_*DM)
  if (t >= B_ * DM) return;
  int b = t >> 9, d = t & (DM - 1);
  float s = 0.f;
  for (int l = 0; l < L_; ++l) {
    size_t row = (size_t)b * L_ + l;
    s += (h[row * DM + d] - stats[row * 2]) * stats[row * 2 + 1];
  }
  pooled[t] = w[d] * (s * (1.f / L_)) + bias[d];
}

// ---------------------------------------------------------------------------
// Classifier head: out[b,c] = pooled[b,:] . cls_w[c,:] + cls_b[c]
// ---------------------------------------------------------------------------
__global__ void k_cls(const float* __restrict__ pooled, const float* __restrict__ cw,
                      const float* __restrict__ cbias, float* __restrict__ out) {
  int t = threadIdx.x;       // 512 threads = 8 batches * 64 classes
  int b = t >> 6, c = t & 63;
  float s = cbias[c];
  for (int d = 0; d < DM; ++d) s += pooled[b * DM + d] * cw[c * DM + d];
  out[b * NC + c] = s;
}

// ---------------------------------------------------------------------------
// Host launcher
// ---------------------------------------------------------------------------
extern "C" void kernel_launch(void* const* d_in, const int* in_sizes, int n_in,
                              void* d_out, int out_size, void* d_ws, size_t ws_size,
                              hipStream_t stream) {
  (void)in_sizes; (void)n_in; (void)out_size; (void)ws_size;

  const int*   x       = (const int*)  d_in[0];
  const float* emb     = (const float*)d_in[1];
  const float* ln_w    = (const float*)d_in[2];
  const float* ln_b    = (const float*)d_in[3];
  const float* in_w    = (const float*)d_in[4];
  const float* conv_w  = (const float*)d_in[5];
  const float* conv_b  = (const float*)d_in[6];
  const float* xp_w    = (const float*)d_in[7];
  const float* dt_w    = (const float*)d_in[8];
  const float* dt_b    = (const float*)d_in[9];
  const float* A_log   = (const float*)d_in[10];
  const float* Dvec    = (const float*)d_in[11];
  const float* out_w   = (const float*)d_in[12];
  const float* norm_w  = (const float*)d_in[13];
  const float* norm_b  = (const float*)d_in[14];
  const float* cls_w   = (const float*)d_in[15];
  const float* cls_b   = (const float*)d_in[16];

  // ---- workspace layout (256-byte aligned slices) ----
  char* ws = (char*)d_ws;
  size_t off = 0;
  auto take = [&](size_t bytes) -> char* {
    char* p = ws + off;
    off += (bytes + 255) & ~(size_t)255;
    return p;
  };
  float*    h      = (float*)   take((size_t)MTOT * DM * 4);
  _Float16* hn16   = (_Float16*)take((size_t)MTOT * DM * 2);
  _Float16* xz16   = (_Float16*)take((size_t)MTOT * 2 * DI * 2);
  _Float16* xc16   = (_Float16*)take((size_t)MTOT * DI * 2);
  float*    xdbl   = (float*)   take((size_t)MTOT * 64 * 4);
  _Float16* dt16   = (_Float16*)take((size_t)MTOT * DTR * 2);
  float*    delta  = (float*)   take((size_t)MTOT * DI * 4);
  _Float16* y16    = (_Float16*)take((size_t)MTOT * DI * 2);
  float*    stats  = (float*)   take((size_t)MTOT * 2 * 4);
  float*    pooled = (float*)   take((size_t)B_ * DM * 4);
  _Float16* w_in16 = (_Float16*)take((size_t)NL * 2 * DI * DM * 2);
  _Float16* w_xp16 = (_Float16*)take((size_t)NL * 64 * DI * 2);
  _Float16* w_dt16 = (_Float16*)take((size_t)NL * DI * DTR * 2);
  _Float16* w_ot16 = (_Float16*)take((size_t)NL * DM * DI * 2);

  auto cvt = [&](const float* s, _Float16* d, int n) {
    k_cvt_f16<<<(n + 255) / 256, 256, 0, stream>>>(s, d, n);
  };
  cvt(in_w,  w_in16, NL * 2 * DI * DM);
  cvt(xp_w,  w_xp16, NL * 64 * DI);
  cvt(dt_w,  w_dt16, NL * DI * DTR);
  cvt(out_w, w_ot16, NL * DM * DI);

  k_embed<<<(MTOT * DM + 255) / 256, 256, 0, stream>>>(x, emb, h);

  // 64x64 tiles, 8 waves per block; all products divisible by 8.
  auto gemm_blocks = [](int M, int N) { return (M / 64) * (N / 64) / 8; };

  for (int layer = 0; layer < NL; ++layer) {
    k_ln_f16<<<MTOT, 256, 0, stream>>>(h, ln_w + layer * DM, ln_b + layer * DM, hn16);

    // in_proj: [MTOT,512] x [2048,512]^T -> xz16 (f16)
    k_gemm_wmma<<<gemm_blocks(MTOT, 2 * DI), 256, 0, stream>>>(
        hn16, w_in16 + (size_t)layer * 2 * DI * DM, xz16, nullptr,
        MTOT, 2 * DI, DM, 2);

    k_conv_silu<<<(MTOT * DI + 255) / 256, 256, 0, stream>>>(
        xz16, conv_w + (size_t)layer * DI * DC, conv_b + (size_t)layer * DI, xc16);

    // x_proj: [MTOT,1024] x [64,1024]^T -> xdbl (f32)
    k_gemm_wmma<<<gemm_blocks(MTOT, 64), 256, 0, stream>>>(
        xc16, w_xp16 + (size_t)layer * 64 * DI, xdbl, nullptr,
        MTOT, 64, DI, 0);

    k_slice_dt<<<(MTOT * DTR + 255) / 256, 256, 0, stream>>>(xdbl, dt16);

    // dt_proj + bias + softplus: [MTOT,32] x [1024,32]^T -> delta (f32)
    k_gemm_wmma<<<gemm_blocks(MTOT, DI), 256, 0, stream>>>(
        dt16, w_dt16 + (size_t)layer * DI * DTR, delta,
        dt_b + (size_t)layer * DI, MTOT, DI, DTR, 3);

    k_scan<<<(B_ * DI) / 256, 256, 0, stream>>>(
        xc16, delta, xdbl, xz16,
        A_log + (size_t)layer * DI * DS, Dvec + (size_t)layer * DI, y16);

    // out_proj with residual add: h += [MTOT,1024] x [512,1024]^T
    k_gemm_wmma<<<gemm_blocks(MTOT, DM), 256, 0, stream>>>(
        y16, w_ot16 + (size_t)layer * DM * DI, h, nullptr,
        MTOT, DM, DI, 1);
  }

  k_lnstats<<<MTOT, 256, 0, stream>>>(h, stats);
  k_pool<<<(B_ * DM + 255) / 256, 256, 0, stream>>>(h, stats, norm_w, norm_b, pooled);
  k_cls<<<1, B_ * NC, 0, stream>>>(pooled, cls_w, cls_b, (float*)d_out);
}
